// InitEPT_48137993453602
// MI455X (gfx1250) — matrix operand order repaired and checked by reading, not compiled, not run on hardware
//
#include <hip/hip_runtime.h>

// ---------------------------------------------------------------------------
// Patch-GEMM for yangEPTNet on MI455X (gfx1250, wave32, WMMA bf16).
//   M = 1024 patch rows (B=2 x 512 patches), N = 4096 channels, K = 4096.
//   acc_r = A_r*W_rr^T + A_i*W_ir^T + (b_rr + b_ir)
//   acc_i = A_i*W_ii^T + A_r*W_ri^T + (b_ii + b_ri)
//
// Fast path (ws >= ~144MB): pack x-patches + weights to bf16 once, then a
// double-buffered LDS GEMM using async global->LDS copies (ASYNCcnt) and
// v_wmma_f32_16x16x32_bf16. Fallback: fused fp32->bf16 staging GEMM.
// ---------------------------------------------------------------------------

typedef __attribute__((ext_vector_type(16))) __bf16         v16bf;
typedef __attribute__((ext_vector_type(8)))  __bf16         v8bf;
typedef __attribute__((ext_vector_type(8)))  float          v8f;
typedef __attribute__((ext_vector_type(4)))  float          v4f;
typedef __attribute__((ext_vector_type(8)))  unsigned short v8u;
typedef __attribute__((ext_vector_type(4)))  int            v4i;
typedef __attribute__((ext_vector_type(4)))  unsigned int   v4u;
typedef __attribute__((ext_vector_type(2)))  unsigned int   v2u;

#define BSZ      16
#define KDIM     4096          // 16^3
#define NCH      4096
#define MROWS    1024          // B * 512 patches
#define TILE_M   128
#define TILE_N   64
#define TILE_K   32
#define LDSROW   40            // row stride in ushorts (80B, 16B aligned, skewed)
#define YI_OFF   4194304       // 2 * 128^3 (offset of y_i inside d_out)
#define AELEMS   4194304       // 1024*4096 bf16 per packed A matrix
#define WELEMS   16777216      // 4096*4096 bf16 per packed W matrix

#if __has_builtin(__builtin_amdgcn_global_load_async_to_lds_b128)
#define USE_ASYNC_LDS 1
#else
#define USE_ASYNC_LDS 0
#endif

#define AS_GLOBAL __attribute__((address_space(1)))
#define AS_SHARED __attribute__((address_space(3)))

static __device__ __forceinline__ unsigned short f2bf(float f) {
    unsigned u = __float_as_uint(f);
    unsigned r = (u + 0x7FFFu + ((u >> 16) & 1u)) >> 16;  // RNE
    return (unsigned short)r;
}

static __device__ __forceinline__ v16bf ld_fragA(const unsigned short* p) {
    // chunks: [k .. k+8), [k+16 .. k+24)  -> two 16B LDS loads
    v8u lo = *(const v8u*)(p);
    v8u hi = *(const v8u*)(p + 16);
    v8bf l = __builtin_bit_cast(v8bf, lo);
    v8bf h = __builtin_bit_cast(v8bf, hi);
    return __builtin_shufflevector(l, h, 0,1,2,3,4,5,6,7,8,9,10,11,12,13,14,15);
}

static __device__ __forceinline__ v16bf ld_fragB(const unsigned short* p) {
    // chunks: [k .. k+8), [k+16 .. k+24)  -> two 16B LDS loads
    v8u lo = *(const v8u*)(p);
    v8u hi = *(const v8u*)(p + 16);
    v8bf l = __builtin_bit_cast(v8bf, lo);
    v8bf h = __builtin_bit_cast(v8bf, hi);
    return __builtin_shufflevector(l, h, 0,1,2,3,4,5,6,7,8,9,10,11,12,13,14,15);
}

static __device__ __forceinline__ v8f wmma_bf16(v16bf a, v16bf b, v8f c) {
    return __builtin_amdgcn_wmma_f32_16x16x32_bf16(
        false, a, false, b, (short)0, c, false, false);
}

static __device__ __forceinline__ void copy16(const unsigned short* g,
                                              unsigned short* l) {
#if USE_ASYNC_LDS
    __builtin_amdgcn_global_load_async_to_lds_b128(
        (AS_GLOBAL v4i*)g, (AS_SHARED v4i*)l, 0, 0);
#else
    v4u v = *(const v4u*)g;
    *(v4u*)l = v;
#endif
}

static __device__ __forceinline__ void wait_async() {
#if USE_ASYNC_LDS
#if __has_builtin(__builtin_amdgcn_s_wait_asynccnt)
    __builtin_amdgcn_s_wait_asynccnt(0);
#else
    asm volatile("s_wait_asynccnt 0x0" ::: "memory");
#endif
#endif
}

// ---------------------------------------------------------------------------
// Pack kernels (fast path)
// ---------------------------------------------------------------------------

// x [B,1,128,128,128] fp32 -> patch-major A [ri][1024][4096] bf16
__global__ __launch_bounds__(256)
void ept_pack_a(const float* __restrict__ xr, const float* __restrict__ xi,
                unsigned short* __restrict__ aPk) {
    int id  = blockIdx.x * 256 + threadIdx.x;     // 0 .. 2*1024*1024-1
    int ri  = id >> 20;
    int rem = id & 1048575;
    int pm  = rem >> 10;                          // patch row
    int k   = (rem & 1023) * 4;                   // float4 chunk
    int b   = pm >> 9, p = pm & 511;
    int hb  = p >> 6, lb = (p >> 3) & 7, db = p & 7;
    int kd  = k >> 8, kh = (k >> 4) & 15, kwc = k & 15;
    const float* __restrict__ xs = ri ? xi : xr;
    v4f v = *(const v4f*)(xs + ((size_t)((b * 128 + hb * BSZ + kd) * 128 +
                                         lb * BSZ + kh) * 128 + db * BSZ + kwc));
    v2u pk;
    pk.x = (unsigned)f2bf(v.x) | ((unsigned)f2bf(v.y) << 16);
    pk.y = (unsigned)f2bf(v.z) | ((unsigned)f2bf(v.w) << 16);
    *(v2u*)(aPk + (size_t)ri * AELEMS + (size_t)pm * KDIM + k) = pk;
}

// W [4][4096][4096] fp32 -> bf16 (layout order: rr, ri, ii, ir)
__global__ __launch_bounds__(256)
void ept_pack_w(const float* __restrict__ Wrr, const float* __restrict__ Wri,
                const float* __restrict__ Wii, const float* __restrict__ Wir,
                unsigned short* __restrict__ wPk) {
    int id  = blockIdx.x * 256 + threadIdx.x;     // 0 .. 16M-1 float4 chunks
    int w   = id >> 22;
    int rem = id & 4194303;
    const float* __restrict__ src = (w == 0) ? Wrr : (w == 1) ? Wri
                                  : (w == 2) ? Wii : Wir;
    v4f v = *(const v4f*)(src + (size_t)rem * 4);
    v2u pk;
    pk.x = (unsigned)f2bf(v.x) | ((unsigned)f2bf(v.y) << 16);
    pk.y = (unsigned)f2bf(v.z) | ((unsigned)f2bf(v.w) << 16);
    *(v2u*)(wPk + (size_t)w * WELEMS + (size_t)rem * 4) = pk;
}

// ---------------------------------------------------------------------------
// Fast-path GEMM: bf16 in, double-buffered LDS, async copies
// ---------------------------------------------------------------------------

struct Tile {
    unsigned short a[2][TILE_M * LDSROW];   // [0]=A_r  [1]=A_i
    unsigned short w[4][TILE_N * LDSROW];   // rr, ri, ii, ir
};

__global__ __launch_bounds__(256)
void ept_gemm_bf16(const unsigned short* __restrict__ aPk,
                   const unsigned short* __restrict__ wPk,
                   const float* __restrict__ brr, const float* __restrict__ bri,
                   const float* __restrict__ bii, const float* __restrict__ bir,
                   float* __restrict__ out) {
    __shared__ __align__(16) Tile sm[2];

    const int tid    = threadIdx.x;
    const int lane   = tid & 31;
    const int wave   = tid >> 5;
    const int waveM  = wave >> 1;
    const int waveN  = wave & 1;
    const int lane15 = lane & 15;
    const int laneHi = lane >> 4;

    const int blockM = blockIdx.x;       // 0..7  (M-major launch: L2 W reuse)
    const int blockN = blockIdx.y;       // 0..63
    const int m0     = blockM * TILE_M;
    const int n0     = blockN * TILE_N;

    float biasR[2], biasI[2];
#pragma unroll
    for (int nt = 0; nt < 2; ++nt) {
        int c = n0 + waveN * 32 + nt * 16 + lane15;
        biasR[nt] = brr[c] + bir[c];
        biasI[nt] = bii[c] + bri[c];
    }

    v8f accR[2][2], accI[2][2];
#pragma unroll
    for (int mt = 0; mt < 2; ++mt)
#pragma unroll
        for (int nt = 0; nt < 2; ++nt) {
            accR[mt][nt] = (v8f)(0.0f);
            accI[mt][nt] = (v8f)(0.0f);
        }

    // Per-thread staging slots: 2048 x 16B chunks per tile, 8 per thread.
    auto stage = [&](int buf, int kb) {
        const int k0 = kb * TILE_K;
#pragma unroll
        for (int it = 0; it < 4; ++it) {
            int idx = tid + it * 256;            // 0..1023
            int ri  = idx >> 9;
            int r2  = idx & 511;
            int row = r2 >> 2;
            int c   = r2 & 3;
            const unsigned short* g = aPk + (size_t)ri * AELEMS +
                                      (size_t)(m0 + row) * KDIM + k0 + c * 8;
            copy16(g, &sm[buf].a[ri][row * LDSROW + c * 8]);
        }
#pragma unroll
        for (int it = 0; it < 4; ++it) {
            int idx = tid + it * 256;            // 0..1023
            int w   = idx >> 8;
            int r2  = idx & 255;
            int row = r2 >> 2;
            int c   = r2 & 3;
            const unsigned short* g = wPk + (size_t)w * WELEMS +
                                      (size_t)(n0 + row) * KDIM + k0 + c * 8;
            copy16(g, &sm[buf].w[w][row * LDSROW + c * 8]);
        }
    };

    stage(0, 0);

    for (int kb = 0; kb < KDIM / TILE_K; ++kb) {
        const int cur = kb & 1;
        wait_async();                 // my copies into sm[cur] are done
        __syncthreads();              // everyone's copies done; prev compute done
        if (kb + 1 < KDIM / TILE_K) stage(cur ^ 1, kb + 1);

        v16bf aF[2][2];               // [ri][mt]
#pragma unroll
        for (int mt = 0; mt < 2; ++mt) {
            int row  = waveM * 32 + mt * 16 + lane15;
            int base = row * LDSROW + laneHi * 8;
            aF[0][mt] = ld_fragA(&sm[cur].a[0][base]);
            aF[1][mt] = ld_fragA(&sm[cur].a[1][base]);
        }
#pragma unroll
        for (int nt = 0; nt < 2; ++nt) {
            int n     = waveN * 32 + nt * 16 + lane15;
            int bbase = n * LDSROW + laneHi * 8;
            v16bf bRR = ld_fragB(&sm[cur].w[0][bbase]);
            v16bf bRI = ld_fragB(&sm[cur].w[1][bbase]);
            v16bf bII = ld_fragB(&sm[cur].w[2][bbase]);
            v16bf bIR = ld_fragB(&sm[cur].w[3][bbase]);
#pragma unroll
            for (int mt = 0; mt < 2; ++mt) {
                accR[mt][nt] = wmma_bf16(aF[0][mt], bRR, accR[mt][nt]);
                accR[mt][nt] = wmma_bf16(aF[1][mt], bIR, accR[mt][nt]);
                accI[mt][nt] = wmma_bf16(aF[1][mt], bII, accI[mt][nt]);
                accI[mt][nt] = wmma_bf16(aF[0][mt], bRI, accI[mt][nt]);
            }
        }
    }

#pragma unroll
    for (int mt = 0; mt < 2; ++mt) {
#pragma unroll
        for (int nt = 0; nt < 2; ++nt) {
            int c  = n0 + waveN * 32 + nt * 16 + lane15;
            int kd = c >> 8, kh = (c >> 4) & 15, kw = c & 15;
            float bR = biasR[nt], bI = biasI[nt];
#pragma unroll
            for (int r = 0; r < 8; ++r) {
                int pm = m0 + waveM * 32 + mt * 16 + laneHi * 8 + r;
                int b  = pm >> 9;
                int p  = pm & 511;
                int hb = p >> 6, lb = (p >> 3) & 7, db = p & 7;
                size_t idx = (size_t)((b * 128 + hb * BSZ + kd) * 128 +
                                      lb * BSZ + kh) * 128 + db * BSZ + kw;
                out[idx]          = accR[mt][nt][r] + bR;
                out[YI_OFF + idx] = accI[mt][nt][r] + bI;
            }
        }
    }
}

// ---------------------------------------------------------------------------
// Fallback: fused fp32 -> bf16 staging GEMM (no workspace required)
// ---------------------------------------------------------------------------

struct SMem {
    unsigned short a[2][TILE_M * LDSROW];
    unsigned short w[4][TILE_N * LDSROW];
};

__global__ __launch_bounds__(256)
void ept_patch_gemm(const float* __restrict__ xr, const float* __restrict__ xi,
                    const float* __restrict__ Wrr, const float* __restrict__ Wri,
                    const float* __restrict__ Wii, const float* __restrict__ Wir,
                    const float* __restrict__ brr, const float* __restrict__ bri,
                    const float* __restrict__ bii, const float* __restrict__ bir,
                    float* __restrict__ out) {
    __shared__ __align__(16) SMem sm;

    const int tid    = threadIdx.x;
    const int lane   = tid & 31;
    const int wave   = tid >> 5;
    const int waveM  = wave >> 1;
    const int waveN  = wave & 1;
    const int lane15 = lane & 15;
    const int laneHi = lane >> 4;

    const int blockM = blockIdx.x;
    const int blockN = blockIdx.y;
    const int n0     = blockN * TILE_N;
    const int m0     = blockM * TILE_M;

    const float* __restrict__ Wmat[4] = { Wrr, Wri, Wii, Wir };

    float biasR[2], biasI[2];
#pragma unroll
    for (int nt = 0; nt < 2; ++nt) {
        int c = n0 + waveN * 32 + nt * 16 + lane15;
        biasR[nt] = brr[c] + bir[c];
        biasI[nt] = bii[c] + bri[c];
    }

    v8f accR[2][2], accI[2][2];
#pragma unroll
    for (int mt = 0; mt < 2; ++mt)
#pragma unroll
        for (int nt = 0; nt < 2; ++nt) {
            accR[mt][nt] = (v8f)(0.0f);
            accI[mt][nt] = (v8f)(0.0f);
        }

    for (int kb = 0; kb < KDIM / TILE_K; ++kb) {
        const int k0  = kb * TILE_K;
        const int kd  = k0 >> 8;
        const int kh0 = (k0 >> 4) & 15;

#pragma unroll
        for (int ri = 0; ri < 2; ++ri) {
            const float* __restrict__ xs = ri ? xi : xr;
#pragma unroll
            for (int it = 0; it < 4; ++it) {
                int t   = tid + it * 256;
                int row = t >> 3;
                int q   = t & 7;
                int seg = q >> 2;
                int kc  = (q & 3) * 4;
                int pm  = m0 + row;
                int b   = pm >> 9;
                int p   = pm & 511;
                int hb  = p >> 6, lb = (p >> 3) & 7, db = p & 7;
                int Hi  = hb * BSZ + kd;
                int Li  = lb * BSZ + kh0 + seg;
                int Di  = db * BSZ + kc;
                v4f v = *(const v4f*)(xs +
                        ((size_t)((b * 128 + Hi) * 128 + Li) * 128 + Di));
                v2u pk;
                pk.x = (unsigned)f2bf(v.x) | ((unsigned)f2bf(v.y) << 16);
                pk.y = (unsigned)f2bf(v.z) | ((unsigned)f2bf(v.w) << 16);
                *(v2u*)(&sm.a[ri][row * LDSROW + seg * 16 + kc]) = pk;
            }
        }
#pragma unroll
        for (int it = 0; it < 8; ++it) {
            int t = tid + it * 256;
            int w = t >> 9;
            int f = t & 511;
            int n = f >> 3;
            int q = f & 7;
            int c = n0 + n;
            v4f v = *(const v4f*)(Wmat[w] + (size_t)c * KDIM + k0 + q * 4);
            v2u pk;
            pk.x = (unsigned)f2bf(v.x) | ((unsigned)f2bf(v.y) << 16);
            pk.y = (unsigned)f2bf(v.z) | ((unsigned)f2bf(v.w) << 16);
            *(v2u*)(&sm.w[w][n * LDSROW + q * 4]) = pk;
        }

        __syncthreads();

        v16bf aF[2][2];
#pragma unroll
        for (int mt = 0; mt < 2; ++mt) {
            int row  = waveM * 32 + mt * 16 + lane15;
            int base = row * LDSROW + laneHi * 8;
            aF[0][mt] = ld_fragA(&sm.a[0][base]);
            aF[1][mt] = ld_fragA(&sm.a[1][base]);
        }
#pragma unroll
        for (int nt = 0; nt < 2; ++nt) {
            int n     = waveN * 32 + nt * 16 + lane15;
            int bbase = n * LDSROW + laneHi * 8;
            v16bf bRR = ld_fragB(&sm.w[0][bbase]);
            v16bf bRI = ld_fragB(&sm.w[1][bbase]);
            v16bf bII = ld_fragB(&sm.w[2][bbase]);
            v16bf bIR = ld_fragB(&sm.w[3][bbase]);
#pragma unroll
            for (int mt = 0; mt < 2; ++mt) {
                accR[mt][nt] = wmma_bf16(aF[0][mt], bRR, accR[mt][nt]);
                accR[mt][nt] = wmma_bf16(aF[1][mt], bIR, accR[mt][nt]);
                accI[mt][nt] = wmma_bf16(aF[1][mt], bII, accI[mt][nt]);
                accI[mt][nt] = wmma_bf16(aF[0][mt], bRI, accI[mt][nt]);
            }
        }

        __syncthreads();
    }

#pragma unroll
    for (int mt = 0; mt < 2; ++mt) {
#pragma unroll
        for (int nt = 0; nt < 2; ++nt) {
            int c  = n0 + waveN * 32 + nt * 16 + lane15;
            int kd = c >> 8, kh = (c >> 4) & 15, kw = c & 15;
            float bR = biasR[nt], bI = biasI[nt];
#pragma unroll
            for (int r = 0; r < 8; ++r) {
                int pm = m0 + waveM * 32 + mt * 16 + laneHi * 8 + r;
                int b  = pm >> 9;
                int p  = pm & 511;
                int hb = p >> 6, lb = (p >> 3) & 7, db = p & 7;
                size_t idx = (size_t)((b * 128 + hb * BSZ + kd) * 128 +
                                      lb * BSZ + kh) * 128 + db * BSZ + kw;
                out[idx]          = accR[mt][nt][r] + bR;
                out[YI_OFF + idx] = accI[mt][nt][r] + bI;
            }
        }
    }
}

// ---------------------------------------------------------------------------

extern "C" void kernel_launch(void* const* d_in, const int* in_sizes, int n_in,
                              void* d_out, int out_size, void* d_ws, size_t ws_size,
                              hipStream_t stream) {
    (void)in_sizes; (void)n_in; (void)out_size;
    const float* xr  = (const float*)d_in[0];
    const float* xi  = (const float*)d_in[1];
    const float* Wrr = (const float*)d_in[2];
    const float* brr = (const float*)d_in[3];
    const float* Wri = (const float*)d_in[4];
    const float* bri = (const float*)d_in[5];
    const float* Wii = (const float*)d_in[6];
    const float* bii = (const float*)d_in[7];
    const float* Wir = (const float*)d_in[8];
    const float* bir = (const float*)d_in[9];
    float* out = (float*)d_out;

    const size_t A_BYTES = (size_t)2 * AELEMS * sizeof(unsigned short); // 16 MB
    const size_t W_BYTES = (size_t)4 * WELEMS * sizeof(unsigned short); // 128 MB

    dim3 block(256, 1, 1);
    dim3 grid(MROWS / TILE_M, NCH / TILE_N, 1);   // 8 x 64, M-major for L2 reuse

    if (d_ws != nullptr && ws_size >= A_BYTES + W_BYTES) {
        unsigned short* aPk = (unsigned short*)d_ws;
        unsigned short* wPk = (unsigned short*)((char*)d_ws + A_BYTES);
        ept_pack_a<<<dim3(8192, 1, 1), block, 0, stream>>>(xr, xi, aPk);
        ept_pack_w<<<dim3(65536, 1, 1), block, 0, stream>>>(Wrr, Wri, Wii, Wir, wPk);
        ept_gemm_bf16<<<grid, block, 0, stream>>>(aPk, wPk, brr, bri, bii, bir, out);
    } else {
        ept_patch_gemm<<<grid, block, 0, stream>>>(xr, xi, Wrr, Wri, Wii, Wir,
                                                   brr, bri, bii, bir, out);
    }
}
